// MultiQueryAttention_8332236554620
// MI455X (gfx1250) — matrix operand order repaired and checked by reading, not compiled, not run on hardware
//
#include <hip/hip_runtime.h>
#include <hip/hip_bf16.h>

// ---------------------------------------------------------------------------
// MQA on gfx1250: f16 WMMA (16x16x32, f32 accum) for all five matmuls.
//  x(f32)->f16, weights transposed to [N][K] f16 (B-fragment friendly)
//  gemm mode 0: q = x@wq      -> q[b][h][n][64]   (f16, pre-permuted)
//  gemm mode 1: kv = x@wkv    -> k[b][n][64], vT[b][64][n] (f16)
//  attention:   flash-style, online softmax, P via LDS (s_wait_dscnt)
//  gemm mode 2: out = ao@wo+b -> f32
// All hot loads are rooted at kernel args (no pointer arrays / PHIs) so they
// lower to global_load_b128, not flat_load (flat ticks LOADcnt *and* DScnt).
// ---------------------------------------------------------------------------

typedef __attribute__((ext_vector_type(16))) _Float16 v16h;
typedef __attribute__((ext_vector_type(8)))  _Float16 v8h;
typedef __attribute__((ext_vector_type(8)))  float    v8f;

#define NB   4
#define SEQ  2048
#define DIM  1024
#define HEADS 16
#define DHEAD 64
#define MTOT (NB * SEQ)          // 8192 rows
#define ATT_SCALE 0.125f         // 64^-0.5, exact in f16

static __device__ inline v8f wmma16(v16h a, v16h b, v8f c) {
    // (neg_a, A, neg_b, B, c_mod, C, reuse_a, reuse_b)
    return __builtin_amdgcn_wmma_f32_16x16x32_f16(false, a, false, b,
                                                  (short)0, c, false, false);
}

// A-fragment (16x32 f16): lane holds row M=lane%16; 16 contiguous K
// starting at K-offset 16*(lane/16).  `p` must already
// include the row base + 16*(lane/16) offset; second chunk is at p+8.
static __device__ inline v16h load_a_frag(const _Float16* p) {
    v8h lo = *(const v8h*)(p);
    v8h hi = *(const v8h*)(p + 8);
    v16h a;
#pragma unroll
    for (int e = 0; e < 8; ++e) { a[e] = lo[e]; a[e + 8] = hi[e]; }
    return a;
}

// ---------------------------------------------------------------------------
__global__ void cvt_f32_f16(const float* __restrict__ in,
                            _Float16* __restrict__ out, int n) {
    int i = blockIdx.x * blockDim.x + threadIdx.x;
    if (i < n) out[i] = (_Float16)in[i];
}

// w[K][Nn] f32 -> wT[Nn][K] f16
__global__ void transpose_cvt(const float* __restrict__ in,
                              _Float16* __restrict__ out, int K, int Nn) {
    int i = blockIdx.x * blockDim.x + threadIdx.x;
    if (i < K * Nn) {
        int kk = i / Nn, n = i - kk * Nn;
        out[(size_t)n * K + kk] = (_Float16)in[i];
    }
}

// ---------------------------------------------------------------------------
// C[M,Nn] = A[M,K] @ Bt[Nn,K]^T.  256 thr = 8 waves (2x4), wave tile 32x64,
// block tile 64(M) x 256(N), K-step 32.  Branch-free inner loop; base
// pointers scalar + indexed by kk so loads stay in the GLOBAL addrspace.
__global__ __launch_bounds__(256)
void gemm_f16(const _Float16* __restrict__ A, const _Float16* __restrict__ Bt,
              void* out0, void* out1, const float* __restrict__ bias,
              int M, int Nn, int K, int mode) {
    const int w    = threadIdx.x >> 5;
    const int lane = threadIdx.x & 31;
    const int lh   = lane >> 4;       // half-wave select
    const int lm   = lane & 15;
    const int M0   = blockIdx.y * 64 + (w >> 2) * 32;
    const int N0   = blockIdx.x * 256 + (w & 3) * 64;

    // scalar base pointers, each a GEP rooted at the kernel arg
    const _Float16* pa0 = A + (size_t)(M0 + lm) * K + 16 * lh;
    const _Float16* pa1 = pa0 + (size_t)16 * K;
    int c0 = N0 + lm;       if (c0 >= Nn) c0 = Nn - 1;   // clamp (kv: Nn=128)
    int c1 = N0 + 16 + lm;  if (c1 >= Nn) c1 = Nn - 1;
    int c2 = N0 + 32 + lm;  if (c2 >= Nn) c2 = Nn - 1;
    int c3 = N0 + 48 + lm;  if (c3 >= Nn) c3 = Nn - 1;
    const _Float16* pb0 = Bt + (size_t)c0 * K + 16 * lh;
    const _Float16* pb1 = Bt + (size_t)c1 * K + 16 * lh;
    const _Float16* pb2 = Bt + (size_t)c2 * K + 16 * lh;
    const _Float16* pb3 = Bt + (size_t)c3 * K + 16 * lh;

    v8f acc[2][4] = {};
    for (int kk = 0; kk < K; kk += 32) {
        v16h a0 = load_a_frag(pa0 + kk);
        v16h a1 = load_a_frag(pa1 + kk);
        v16h b0 = *(const v16h*)(pb0 + kk);
        v16h b1 = *(const v16h*)(pb1 + kk);
        v16h b2 = *(const v16h*)(pb2 + kk);
        v16h b3 = *(const v16h*)(pb3 + kk);
        acc[0][0] = wmma16(a0, b0, acc[0][0]);
        acc[0][1] = wmma16(a0, b1, acc[0][1]);
        acc[0][2] = wmma16(a0, b2, acc[0][2]);
        acc[0][3] = wmma16(a0, b3, acc[0][3]);
        acc[1][0] = wmma16(a1, b0, acc[1][0]);
        acc[1][1] = wmma16(a1, b1, acc[1][1]);
        acc[1][2] = wmma16(a1, b2, acc[1][2]);
        acc[1][3] = wmma16(a1, b3, acc[1][3]);
    }

    // bias values depend only on Ncol (ni,lm) -> load once for mode 2
    float bv[4] = {0.f, 0.f, 0.f, 0.f};
    if (mode == 2) {
#pragma unroll
        for (int ni = 0; ni < 4; ++ni) bv[ni] = bias[N0 + ni * 16 + lm];
    }

    // D layout: vgpr r -> (M = r + 8*lh, N = lm) within each 16x16 tile
#pragma unroll
    for (int mi = 0; mi < 2; ++mi)
#pragma unroll
    for (int ni = 0; ni < 4; ++ni)
#pragma unroll
    for (int r = 0; r < 8; ++r) {
        int Mrow = M0 + mi * 16 + r + 8 * lh;
        int Ncol = N0 + ni * 16 + lm;
        if (Ncol >= Nn) continue;
        float val = acc[mi][ni][r];
        int bi = Mrow >> 11;          // / SEQ
        int nr = Mrow & (SEQ - 1);
        if (mode == 0) {              // q: [b][h][n][64]
            int hh = Ncol >> 6, dh = Ncol & 63;
            ((_Float16*)out0)[(((size_t)bi * HEADS + hh) * SEQ + nr) * DHEAD + dh] =
                (_Float16)val;
        } else if (mode == 1) {       // kv: k [b][n][64], vT [b][64][n]
            if (Ncol < DHEAD)
                ((_Float16*)out0)[((size_t)bi * SEQ + nr) * DHEAD + Ncol] = (_Float16)val;
            else
                ((_Float16*)out1)[((size_t)bi * DHEAD + (Ncol - DHEAD)) * SEQ + nr] =
                    (_Float16)val;
        } else {                      // out-proj: f32 + bias
            ((float*)out0)[(size_t)Mrow * Nn + Ncol] = val + bv[ni];
        }
    }
}

// ---------------------------------------------------------------------------
// Flash attention, one (b, h, 64-query block) per 128-thread block (4 waves,
// 16 queries/wave).  K loop in 32-key steps: 4 QK^T wmma, online softmax
// with 16-lane xor reductions, P through LDS into A-layout, 4 PV wmma.
__global__ __launch_bounds__(128)
void attn_kernel(const _Float16* __restrict__ q, const _Float16* __restrict__ k,
                 const _Float16* __restrict__ vt, _Float16* __restrict__ ao) {
    const int w    = threadIdx.x >> 5;
    const int lane = threadIdx.x & 31;
    const int lh   = lane >> 4;
    const int lm   = lane & 15;
    const int q0   = blockIdx.x * 64 + w * 16;
    const int hh   = blockIdx.y;
    const int bb   = blockIdx.z;

    __shared__ __align__(16) _Float16 Pbuf[4][16][48];   // 16x32 P + pad

    // Q fragments (prescaled by 1/sqrt(dh)), dh 0..31 and 32..63
    const _Float16* qrow =
        q + (((size_t)(bb * HEADS + hh) * SEQ) + q0 + lm) * DHEAD + 16 * lh;
    v16h aq[2];
#pragma unroll
    for (int j = 0; j < 2; ++j) {
        aq[j] = load_a_frag(qrow + j * 32);
#pragma unroll
        for (int e = 0; e < 16; ++e) aq[j][e] *= (_Float16)ATT_SCALE;
    }

    v8f o[4] = {};
    float mrow[8], lrow[8];
#pragma unroll
    for (int r = 0; r < 8; ++r) { mrow[r] = -3.0e38f; lrow[r] = 0.f; }

    // scalar K / V base pointers rooted at kernel args (global_load)
    const _Float16* kp0 = k + (size_t)bb * SEQ * DHEAD + (size_t)lm * DHEAD + 16 * lh;
    const _Float16* kp1 = kp0 + 16 * DHEAD;
    const _Float16* vbase = vt + (size_t)bb * DHEAD * SEQ + (size_t)lm * SEQ + 16 * lh;
    const _Float16* vp0 = vbase;
    const _Float16* vp1 = vbase + (size_t)16 * SEQ;
    const _Float16* vp2 = vbase + (size_t)32 * SEQ;
    const _Float16* vp3 = vbase + (size_t)48 * SEQ;
    const _Float16* prow = &Pbuf[w][lm][16 * lh];

    for (int n0 = 0; n0 < SEQ; n0 += 32) {
        const int ko = n0 * DHEAD;
        // S = Q K^T : two 16x16 tiles, K-dim = dh = 64 -> 2 wmma each
        v8f s[2];
        {
            v8f accu = {};
            accu = wmma16(aq[0], *(const v16h*)(kp0 + ko),      accu);
            accu = wmma16(aq[1], *(const v16h*)(kp0 + ko + 32), accu);
            s[0] = accu;
        }
        {
            v8f accu = {};
            accu = wmma16(aq[0], *(const v16h*)(kp1 + ko),      accu);
            accu = wmma16(aq[1], *(const v16h*)(kp1 + ko + 32), accu);
            s[1] = accu;
        }

        // online softmax; each D-row lives across one 16-lane half-wave
#pragma unroll
        for (int r = 0; r < 8; ++r) {
            float v = fmaxf(s[0][r], s[1][r]);
#pragma unroll
            for (int off = 1; off < 16; off <<= 1) v = fmaxf(v, __shfl_xor(v, off, 32));
            float nm    = fmaxf(mrow[r], v);
            float alpha = __expf(mrow[r] - nm);
            mrow[r] = nm;
            float p0 = __expf(s[0][r] - nm);
            float p1 = __expf(s[1][r] - nm);
            s[0][r] = p0; s[1][r] = p1;
            float rs = p0 + p1;
#pragma unroll
            for (int off = 1; off < 16; off <<= 1) rs += __shfl_xor(rs, off, 32);
            lrow[r] = lrow[r] * alpha + rs;
#pragma unroll
            for (int di = 0; di < 4; ++di) o[di][r] *= alpha;
        }

        // P (D-layout) -> LDS row-major [16 q][32 keys]
#pragma unroll
        for (int t = 0; t < 2; ++t)
#pragma unroll
            for (int r = 0; r < 8; ++r)
                Pbuf[w][r + 8 * lh][t * 16 + lm] = (_Float16)s[t][r];
        asm volatile("s_wait_dscnt 0" ::: "memory");      // CDNA5 split DS counter

        // reload P in A-fragment layout (16x32, K = 32 keys)
        v16h ap = load_a_frag(prow);

        // O += P @ V : vT is [dh][n] so B-fragments are contiguous
        o[0] = wmma16(ap, *(const v16h*)(vp0 + n0), o[0]);
        o[1] = wmma16(ap, *(const v16h*)(vp1 + n0), o[1]);
        o[2] = wmma16(ap, *(const v16h*)(vp2 + n0), o[2]);
        o[3] = wmma16(ap, *(const v16h*)(vp3 + n0), o[3]);
    }

    // epilogue: normalize and scatter to ao[b][n][h*64+dh] (f16)
#pragma unroll
    for (int r = 0; r < 8; ++r) {
        float inv = 1.0f / lrow[r];
        int nr = q0 + r + 8 * lh;
        size_t base = ((size_t)bb * SEQ + nr) * DIM + hh * DHEAD;
#pragma unroll
        for (int di = 0; di < 4; ++di)
            ao[base + di * 16 + lm] = (_Float16)(o[di][r] * inv);
    }
}

// ---------------------------------------------------------------------------
extern "C" void kernel_launch(void* const* d_in, const int* in_sizes, int n_in,
                              void* d_out, int out_size, void* d_ws, size_t ws_size,
                              hipStream_t stream) {
    const float* x     = (const float*)d_in[0];
    const float* w_q   = (const float*)d_in[1];
    const float* w_kv  = (const float*)d_in[2];
    const float* w_out = (const float*)d_in[3];
    const float* b_out = (const float*)d_in[4];
    float* out = (float*)d_out;

    char* ws = (char*)d_ws;
    size_t oXH  = 0;
    size_t oWQ  = oXH  + (size_t)MTOT * DIM * 2;          // x f16       16 MB
    size_t oWKV = oWQ  + (size_t)DIM * DIM * 2;           // wqT f16      2 MB
    size_t oWO  = oWKV + (size_t)(2 * DHEAD) * DIM * 2;   // wkvT f16   256 KB
    size_t oQ   = oWO  + (size_t)DIM * DIM * 2;           // woT f16      2 MB
    size_t oK   = oQ   + (size_t)MTOT * DIM * 2;          // q f16       16 MB
    size_t oVT  = oK   + (size_t)NB * SEQ * DHEAD * 2;    // k f16        1 MB
    size_t oAO  = oVT  + (size_t)NB * DHEAD * SEQ * 2;    // vT f16       1 MB
    _Float16* xh  = (_Float16*)(ws + oXH);
    _Float16* wqT = (_Float16*)(ws + oWQ);
    _Float16* wkT = (_Float16*)(ws + oWKV);
    _Float16* woT = (_Float16*)(ws + oWO);
    _Float16* qb  = (_Float16*)(ws + oQ);
    _Float16* kb  = (_Float16*)(ws + oK);
    _Float16* vtb = (_Float16*)(ws + oVT);
    _Float16* aob = (_Float16*)(ws + oAO);

    // 1) convert / transpose to f16
    {
        int n = MTOT * DIM;
        cvt_f32_f16<<<(n + 255) / 256, 256, 0, stream>>>(x, xh, n);
        int nq = DIM * DIM;
        transpose_cvt<<<(nq + 255) / 256, 256, 0, stream>>>(w_q, wqT, DIM, DIM);
        int nk = DIM * 2 * DHEAD;
        transpose_cvt<<<(nk + 255) / 256, 256, 0, stream>>>(w_kv, wkT, DIM, 2 * DHEAD);
        transpose_cvt<<<(nq + 255) / 256, 256, 0, stream>>>(w_out, woT, DIM, DIM);
    }
    // 2) q / kv projections
    gemm_f16<<<dim3(DIM / 256, MTOT / 64), 256, 0, stream>>>(
        xh, wqT, qb, nullptr, nullptr, MTOT, DIM, DIM, 0);
    gemm_f16<<<dim3(1, MTOT / 64), 256, 0, stream>>>(
        xh, wkT, kb, vtb, nullptr, MTOT, 2 * DHEAD, DIM, 1);
    // 3) attention
    attn_kernel<<<dim3(SEQ / 64, HEADS, NB), 128, 0, stream>>>(qb, kb, vtb, aob);
    // 4) output projection + bias (f32 out)
    gemm_f16<<<dim3(DIM / 256, MTOT / 64), 256, 0, stream>>>(
        aob, woT, out, nullptr, b_out, MTOT, DIM, DIM, 2);
}
